// GCNWithEdgeFeatures_33509334843816
// MI455X (gfx1250) — matrix operand order repaired and checked by reading, not compiled, not run on hardware
//
#include <hip/hip_runtime.h>

typedef __attribute__((ext_vector_type(16))) _Float16 v16h;
typedef __attribute__((ext_vector_type(8)))  _Float16 v8h;
typedef __attribute__((ext_vector_type(8)))  float    v8f;
typedef __attribute__((ext_vector_type(4)))  int      v4i;

#define NE 100000   // edges
#define NV 50000    // nodes
#define NG 2000     // graphs

// ---------------------------------------------------------------------------
// CDNA5 async copy global->LDS (ASYNCcnt-tracked), 16B per lane per issue
// ---------------------------------------------------------------------------
__device__ __forceinline__ void async_copy16(const _Float16* g, _Float16* l) {
#if __has_builtin(__builtin_amdgcn_global_load_async_to_lds_b128)
  __builtin_amdgcn_global_load_async_to_lds_b128(
      (__attribute__((address_space(1))) v4i*)g,
      (__attribute__((address_space(3))) v4i*)l, 0, 0);
#else
  unsigned loff = (unsigned)(unsigned long long)
      ((__attribute__((address_space(3))) void*)l);
  asm volatile("global_load_async_to_lds_b128 %0, %1, off"
               :: "v"(loff), "v"(g) : "memory");
#endif
}

template <int N>
__device__ __forceinline__ void wait_async() {
#if __has_builtin(__builtin_amdgcn_s_wait_asynccnt)
  __builtin_amdgcn_s_wait_asynccnt(N);
#else
  asm volatile("s_wait_asynccnt %0" :: "i"(N) : "memory");
#endif
}

// ---------------------------------------------------------------------------
// utility kernels
// ---------------------------------------------------------------------------
__global__ void zero_k(int* __restrict__ p, long n) {
  long i = (long)blockIdx.x * blockDim.x + threadIdx.x;
  if (i < n) p[i] = 0;
}

// W[Kin][N] (fp32, row-major) -> Wt[N][Kpad] (f16, zero-padded K)
__global__ void cvt_t_k(const float* __restrict__ W, _Float16* __restrict__ Wt,
                        int Kin, int N, int Kpad) {
  long i = (long)blockIdx.x * blockDim.x + threadIdx.x;
  long total = (long)N * Kpad;
  if (i >= total) return;
  int n = (int)(i / Kpad);
  int k = (int)(i % Kpad);
  float v = (k < Kin) ? W[(size_t)k * N + n] : 0.0f;
  Wt[i] = (_Float16)v;
}

// MSB-first bit unpack of two bytes -> ea16[E][32] (cols 16..31 zero-padded)
__global__ void unpack_k(const int* __restrict__ eap, _Float16* __restrict__ ea) {
  long i = (long)blockIdx.x * blockDim.x + threadIdx.x;
  if (i >= (long)NE * 32) return;
  int e = (int)(i >> 5);
  int c = (int)(i & 31);
  float v = 0.0f;
  if (c < 16) {
    int w = c >> 3, j = c & 7;
    int byte = eap[e * 2 + w];
    v = (float)((byte >> (7 - j)) & 1);
  }
  ea[i] = (_Float16)v;
}

// in-degree per node + node count per graph
__global__ void degree_k(const int* __restrict__ ei, int* __restrict__ deg,
                         const int* __restrict__ batch, int* __restrict__ gcount) {
  int i = blockIdx.x * blockDim.x + threadIdx.x;
  if (i < NE) atomicAdd(&deg[ei[NE + i]], 1);
  if (i < NV) atomicAdd(&gcount[batch[i]], 1);
}

// ---------------------------------------------------------------------------
// WMMA GEMM: D[M][N] = act(A[M][K] @ Bt[N][K]^T + bias[N]), f16 in, f32 acc,
// f16 out. Block = 128 threads (4 waves), tile = 64(M) x 64(N); B tile is
// double-buffered in LDS via async global->LDS copies and shared by all 4
// waves. K mult of 32, N mult of 64; M arbitrary (clamp + guard).
// grid = (N/64, ceil(M/64)), block = 128.
// ---------------------------------------------------------------------------
#define BROW 40   // LDS row stride in halves (80B: 16B-aligned, bank-spread)

__device__ __forceinline__ v16h load_a_frag(const _Float16* __restrict__ a,
                                            int ld, int m0, int k0, int lane,
                                            int M) {
  int row = m0 + (lane & 15);
  row = row < M ? row : M - 1;                      // clamp (EXEC stays all-1)
  const int kb = (lane >> 4) << 3;                  // 0 or 8
  const _Float16* p = a + (size_t)row * ld + k0 + kb;
  v8h lo = *(const v8h*)(p);                        // K = kb+0..7
  v8h hi = *(const v8h*)(p + 16);                   // K = kb+16..23
  return __builtin_shufflevector(lo, hi, 0, 1, 2, 3, 4, 5, 6, 7,
                                 8, 9, 10, 11, 12, 13, 14, 15);
}

__device__ __forceinline__ v16h lds_b_frag(const _Float16* s, int t, int lane) {
  const int col = lane & 15;
  const int kb  = (lane >> 4) << 3;                 // 0 or 8
  const _Float16* p = s + (t * 16 + col) * BROW + kb;
  v8h lo = *(const v8h*)(p);                        // K = kb+0..7
  v8h hi = *(const v8h*)(p + 16);                   // K = kb+16..23
  return __builtin_shufflevector(lo, hi, 0, 1, 2, 3, 4, 5, 6, 7,
                                 8, 9, 10, 11, 12, 13, 14, 15);
}

// stage 64(N) x 32(K) f16 tile of Bt into LDS: 2 async b128 issues per thread
__device__ __forceinline__ void stage_b(const _Float16* __restrict__ Bt, int ldb,
                                        int n0, int k0, _Float16* dst, int tid) {
#pragma unroll
  for (int j = 0; j < 2; ++j) {
    int idx = j * 128 + tid;                        // 0..255 (16B chunks)
    int c   = idx >> 2;                             // 0..63  (N column)
    int ch  = (idx & 3) << 3;                       // half offset 0/8/16/24
    async_copy16(Bt + (size_t)(n0 + c) * ldb + k0 + ch, dst + c * BROW + ch);
  }
}

__global__ void __launch_bounds__(128)
wmma_gemm_k(const _Float16* __restrict__ A, int lda,
            const _Float16* __restrict__ Bt, int ldb,
            const float* __restrict__ bias,
            _Float16* __restrict__ D, int ldd,
            int M, int K, int relu) {
  __shared__ _Float16 Bsh[2][64 * BROW];
  const int tid  = threadIdx.x;
  const int lane = tid & 31;
  const int wv   = tid >> 5;
  const int n0 = blockIdx.x * 64;
  const int m0 = blockIdx.y * 64 + wv * 16;

  stage_b(Bt, ldb, n0, 0, &Bsh[0][0], tid);

  v8f acc[4] = {};
  int buf = 0;
  for (int k0 = 0; k0 < K; k0 += 32) {
    if (k0 + 32 < K) {
      stage_b(Bt, ldb, n0, k0 + 32, &Bsh[buf ^ 1][0], tid);
      wait_async<2>();                              // prev stage done
    } else {
      wait_async<0>();
    }
    __syncthreads();
    v16h a = load_a_frag(A, lda, m0, k0, lane, M);
#pragma unroll
    for (int t = 0; t < 4; ++t) {
      v16h b = lds_b_frag(&Bsh[buf][0], t, lane);
      acc[t] = __builtin_amdgcn_wmma_f32_16x16x32_f16(
          false, a, false, b, (short)0, acc[t], false, false);
    }
    __syncthreads();                                // reads done before reuse
    buf ^= 1;
  }

  const int col = lane & 15;
  const int rbase = (lane >> 4) << 3;               // 0 or 8
#pragma unroll
  for (int t = 0; t < 4; ++t) {
    float bv = bias[n0 + t * 16 + col];
#pragma unroll
    for (int r = 0; r < 8; ++r) {
      int row = m0 + rbase + r;
      if (row < M) {
        float v = acc[t][r] + bv;
        if (relu) v = v > 0.0f ? v : 0.0f;
        D[(size_t)row * ldd + n0 + t * 16 + col] = (_Float16)v;
      }
    }
  }
}

// ---------------------------------------------------------------------------
// per-edge matmul + scatter-add (conv1: 32->32). one wave per edge.
// ---------------------------------------------------------------------------
__global__ void msg1_k(const float* __restrict__ x, const int* __restrict__ ei,
                       const _Float16* __restrict__ theta,
                       float* __restrict__ agg) {
  int e = blockIdx.x * 8 + (threadIdx.x >> 5);
  if (e >= NE) return;
  int lane = threadIdx.x & 31;
  const float* xv = x + (size_t)ei[e] * 32;
  const _Float16* th = theta + (size_t)e * 1024 + lane;
  float s = 0.0f;
#pragma unroll
  for (int i = 0; i < 32; ++i) s += xv[i] * (float)th[i * 32];
  atomicAdd(&agg[(size_t)ei[NE + e] * 32 + lane], s);
}

// h[v] = relu(agg/deg + x@root1 + bias1). one wave per node.
__global__ void nodeupd1_k(const float* __restrict__ x,
                           const float* __restrict__ agg,
                           const int* __restrict__ deg,
                           const float* __restrict__ root1,
                           const float* __restrict__ bias1,
                           float* __restrict__ h) {
  int v = blockIdx.x * 8 + (threadIdx.x >> 5);
  if (v >= NV) return;
  int lane = threadIdx.x & 31;
  const float* xv = x + (size_t)v * 32;
  float r = 0.0f;
#pragma unroll
  for (int i = 0; i < 32; ++i) r += xv[i] * root1[i * 32 + lane];
  float d = (float)max(deg[v], 1);
  float val = agg[(size_t)v * 32 + lane] / d + r + bias1[lane];
  h[(size_t)v * 32 + lane] = val > 0.0f ? val : 0.0f;
}

// per-edge matmul + scatter-add (conv2: 32->64). one wave per edge, 2 outs/lane.
__global__ void msg2_k(const float* __restrict__ h, const int* __restrict__ ei,
                       const _Float16* __restrict__ theta,
                       float* __restrict__ agg) {
  int e = blockIdx.x * 8 + (threadIdx.x >> 5);
  if (e >= NE) return;
  int lane = threadIdx.x & 31;
  const float* hv = h + (size_t)ei[e] * 32;
  const _Float16* th = theta + (size_t)e * 2048;
  float s0 = 0.0f, s1 = 0.0f;
#pragma unroll
  for (int i = 0; i < 32; ++i) {
    float xi = hv[i];
    s0 += xi * (float)th[i * 64 + lane];
    s1 += xi * (float)th[i * 64 + lane + 32];
  }
  size_t o = (size_t)ei[NE + e] * 64 + lane;
  atomicAdd(&agg[o], s0);
  atomicAdd(&agg[o + 32], s1);
}

// h2 = relu(agg/deg + h@root2 + bias2); accumulate into per-graph sums.
__global__ void nodeupd2_k(const float* __restrict__ h,
                           const float* __restrict__ agg,
                           const int* __restrict__ deg,
                           const float* __restrict__ root2,
                           const float* __restrict__ bias2,
                           const int* __restrict__ batch,
                           float* __restrict__ gsum) {
  int v = blockIdx.x * 8 + (threadIdx.x >> 5);
  if (v >= NV) return;
  int lane = threadIdx.x & 31;
  const float* hv = h + (size_t)v * 32;
  float r0 = 0.0f, r1 = 0.0f;
#pragma unroll
  for (int i = 0; i < 32; ++i) {
    float xi = hv[i];
    r0 += xi * root2[i * 64 + lane];
    r1 += xi * root2[i * 64 + lane + 32];
  }
  float d = (float)max(deg[v], 1);
  float o0 = agg[(size_t)v * 64 + lane] / d + r0 + bias2[lane];
  float o1 = agg[(size_t)v * 64 + lane + 32] / d + r1 + bias2[lane + 32];
  o0 = o0 > 0.0f ? o0 : 0.0f;
  o1 = o1 > 0.0f ? o1 : 0.0f;
  size_t g = (size_t)batch[v] * 64 + lane;
  atomicAdd(&gsum[g], o0);
  atomicAdd(&gsum[g + 32], o1);
}

// g0[g][j] = gsum/count -> f16
__global__ void pool_k(const float* __restrict__ gsum,
                       const int* __restrict__ gcount,
                       _Float16* __restrict__ g0) {
  int i = blockIdx.x * blockDim.x + threadIdx.x;
  if (i >= NG * 64) return;
  int g = i >> 6;
  float c = (float)max(gcount[g], 1);
  g0[i] = (_Float16)(gsum[i] / c);
}

// final tiny projection: [2000,256] @ [256,3] + b -> d_out (fp32)
__global__ void fc4_k(const _Float16* __restrict__ g3,
                      const float* __restrict__ W4, const float* __restrict__ b4,
                      float* __restrict__ out) {
  int i = blockIdx.x * blockDim.x + threadIdx.x;
  if (i >= NG * 3) return;
  int g = i / 3, c = i % 3;
  const _Float16* gv = g3 + (size_t)g * 256;
  float s = b4[c];
  for (int j = 0; j < 256; ++j) s += (float)gv[j] * W4[j * 3 + c];
  out[i] = s;
}

// ---------------------------------------------------------------------------
// host launch
// ---------------------------------------------------------------------------
extern "C" void kernel_launch(void* const* d_in, const int* in_sizes, int n_in,
                              void* d_out, int out_size, void* d_ws, size_t ws_size,
                              hipStream_t stream) {
  const float* x     = (const float*)d_in[0];
  const int*   ei    = (const int*)d_in[1];
  const int*   eap   = (const int*)d_in[2];
  const int*   batch = (const int*)d_in[3];
  const float* W1a = (const float*)d_in[4];  const float* b1a = (const float*)d_in[5];
  const float* W1b = (const float*)d_in[6];  const float* b1b = (const float*)d_in[7];
  const float* W2a = (const float*)d_in[8];  const float* b2a = (const float*)d_in[9];
  const float* W2b = (const float*)d_in[10]; const float* b2b = (const float*)d_in[11];
  const float* root1 = (const float*)d_in[12]; const float* bias1 = (const float*)d_in[13];
  const float* root2 = (const float*)d_in[14]; const float* bias2 = (const float*)d_in[15];
  const float* fcW1 = (const float*)d_in[16]; const float* fcb1 = (const float*)d_in[17];
  const float* fcW2 = (const float*)d_in[18]; const float* fcb2 = (const float*)d_in[19];
  const float* fcW3 = (const float*)d_in[20]; const float* fcb3 = (const float*)d_in[21];
  const float* fcW4 = (const float*)d_in[22]; const float* fcb4 = (const float*)d_in[23];
  float* out = (float*)d_out;
  char* ws = (char*)d_ws;

  size_t off = 0;
  auto take = [&](size_t bytes) -> size_t {
    size_t o = off; off += (bytes + 255) & ~(size_t)255; return o;
  };
  const size_t o_ea   = take((size_t)NE * 32 * 2);
  const size_t o_w1at = take((size_t)1024 * 32 * 2);
  const size_t o_w1bt = take((size_t)1024 * 1024 * 2);
  const size_t o_w2at = take((size_t)1024 * 32 * 2);
  const size_t o_w2bt = take((size_t)2048 * 1024 * 2);
  const size_t o_f1t  = take((size_t)512 * 64 * 2);
  const size_t o_f2t  = take((size_t)512 * 512 * 2);
  const size_t o_f3t  = take((size_t)256 * 512 * 2);
  const size_t o_hid  = take((size_t)NE * 1024 * 2);   // hid1 then hid2
  const size_t o_th   = take((size_t)NE * 2048 * 2);   // theta1 then theta2
  const long  ZW      = (long)NV + NG + (long)NV * 32 + (long)NV * 64 + (long)NG * 64;
  const size_t o_zero = take((size_t)ZW * 4);
  const size_t o_h    = take((size_t)NV * 32 * 4);
  const size_t o_g0   = take((size_t)NG * 64 * 2);
  const size_t o_g1   = take((size_t)NG * 512 * 2);
  const size_t o_g2   = take((size_t)NG * 512 * 2);
  const size_t o_g3   = take((size_t)NG * 256 * 2);
  (void)ws_size; (void)in_sizes; (void)n_in; (void)out_size;

  _Float16* ea16  = (_Float16*)(ws + o_ea);
  _Float16* w1at  = (_Float16*)(ws + o_w1at);
  _Float16* w1bt  = (_Float16*)(ws + o_w1bt);
  _Float16* w2at  = (_Float16*)(ws + o_w2at);
  _Float16* w2bt  = (_Float16*)(ws + o_w2bt);
  _Float16* f1t   = (_Float16*)(ws + o_f1t);
  _Float16* f2t   = (_Float16*)(ws + o_f2t);
  _Float16* f3t   = (_Float16*)(ws + o_f3t);
  _Float16* hid   = (_Float16*)(ws + o_hid);
  _Float16* theta = (_Float16*)(ws + o_th);
  int*   deg    = (int*)(ws + o_zero);
  int*   gcount = deg + NV;
  float* agg1   = (float*)(gcount + NG);
  float* agg2   = agg1 + (size_t)NV * 32;
  float* gsum   = agg2 + (size_t)NV * 64;
  float* h      = (float*)(ws + o_h);
  _Float16* g0  = (_Float16*)(ws + o_g0);
  _Float16* g1  = (_Float16*)(ws + o_g1);
  _Float16* g2  = (_Float16*)(ws + o_g2);
  _Float16* g3  = (_Float16*)(ws + o_g3);

  // 0) zero accumulators (harness does not re-zero between replays)
  zero_k<<<dim3((unsigned)((ZW + 255) / 256)), dim3(256), 0, stream>>>(deg, ZW);

  // 1) weight transpose + f16 conversion (B operands stored N-major)
  auto cvt = [&](const float* W, _Float16* Wt, int Kin, int N, int Kpad) {
    long tot = (long)N * Kpad;
    cvt_t_k<<<dim3((unsigned)((tot + 255) / 256)), dim3(256), 0, stream>>>(W, Wt, Kin, N, Kpad);
  };
  cvt(W1a, w1at, 16, 1024, 32);
  cvt(W1b, w1bt, 1024, 1024, 1024);
  cvt(W2a, w2at, 16, 1024, 32);
  cvt(W2b, w2bt, 1024, 2048, 1024);
  cvt(fcW1, f1t, 64, 512, 64);
  cvt(fcW2, f2t, 512, 512, 512);
  cvt(fcW3, f3t, 512, 256, 512);

  // 2) edge feature bit-unpack (f16, K padded to 32)
  unpack_k<<<dim3((NE * 32 + 255) / 256), dim3(256), 0, stream>>>(eap, ea16);

  // 3) degrees + graph node counts
  degree_k<<<dim3((NE + 255) / 256), dim3(256), 0, stream>>>(ei, deg, batch, gcount);

  auto gemm = [&](const _Float16* A, int lda, const _Float16* Bt, int ldb,
                  const float* bias, _Float16* D, int ldd, int M, int N, int K,
                  int relu) {
    wmma_gemm_k<<<dim3((unsigned)(N / 64), (unsigned)((M + 63) / 64)),
                  dim3(128), 0, stream>>>(A, lda, Bt, ldb, bias, D, ldd, M, K, relu);
  };

  // 4) theta1 = relu(ea@W1a+b1a)@W1b + b1b   (the 210 GFLOP path)
  gemm(ea16, 32, w1at, 32, b1a, hid, 1024, NE, 1024, 32, 1);
  gemm(hid, 1024, w1bt, 1024, b1b, theta, 1024, NE, 1024, 1024, 0);

  // 5) conv1 message passing + node update
  msg1_k<<<dim3((NE + 7) / 8), dim3(256), 0, stream>>>(x, ei, theta, agg1);
  nodeupd1_k<<<dim3((NV + 7) / 8), dim3(256), 0, stream>>>(x, agg1, deg, root1, bias1, h);

  // 6) theta2 = relu(ea@W2a+b2a)@W2b + b2b   (the 420 GFLOP path; reuses buffers)
  gemm(ea16, 32, w2at, 32, b2a, hid, 1024, NE, 1024, 32, 1);
  gemm(hid, 1024, w2bt, 1024, b2b, theta, 2048, NE, 2048, 1024, 0);

  // 7) conv2 message passing + node update + graph pooling accumulate
  msg2_k<<<dim3((NE + 7) / 8), dim3(256), 0, stream>>>(h, ei, theta, agg2);
  nodeupd2_k<<<dim3((NV + 7) / 8), dim3(256), 0, stream>>>(h, agg2, deg, root2, bias2, batch, gsum);
  pool_k<<<dim3((NG * 64 + 255) / 256), dim3(256), 0, stream>>>(gsum, gcount, g0);

  // 8) FC stack on [2000, ...] via WMMA
  gemm(g0, 64, f1t, 64, fcb1, g1, 512, NG, 512, 64, 1);
  gemm(g1, 512, f2t, 512, fcb2, g2, 512, NG, 512, 512, 1);
  gemm(g2, 512, f3t, 512, fcb3, g3, 256, NG, 256, 512, 1);

  // 9) final 256->3 projection (fp32)
  fc4_k<<<dim3((NG * 3 + 255) / 256), dim3(256), 0, stream>>>(g3, fcW4, fcb4, out);
}
